// _CausalSelfAttention_23519240913017
// MI455X (gfx1250) — matrix operand order repaired and checked by reading, not compiled, not run on hardware
//
#include <hip/hip_runtime.h>

typedef __bf16 bf16_t;
typedef __attribute__((ext_vector_type(16))) __bf16 v16bf;
typedef __attribute__((ext_vector_type(8)))  __bf16 v8bf;
typedef __attribute__((ext_vector_type(8)))  float  v8f;
typedef __attribute__((ext_vector_type(4)))  int    v4i;

#define GLOBAL_AS __attribute__((address_space(1)))
#define LDS_AS    __attribute__((address_space(3)))

#if defined(__AMDGCN__) && __has_builtin(__builtin_amdgcn_global_load_async_to_lds_b128)
#define ASYNC_LDS 1
#else
#define ASYNC_LDS 0
#endif

// Copy 16B (8 x bf16) global -> LDS. On gfx1250 this is the ASYNCcnt-tracked
// async path (no VGPR round trip); fallback is a plain LDS store.
// Probe-derived signature: (v4i addrspace(1)* src, v4i addrspace(3)* dst,
//                           imm offset, imm cpol)
__device__ __forceinline__ void async_ld16(void* lds, const void* g) {
#if ASYNC_LDS
  __builtin_amdgcn_global_load_async_to_lds_b128(
      (GLOBAL_AS v4i*)const_cast<void*>(g), (LDS_AS v4i*)lds, 0, 0);
#else
  *(v8bf*)lds = *(const v8bf*)g;
#endif
}

__device__ __forceinline__ void wait_async() {
#if ASYNC_LDS
#if __has_builtin(__builtin_amdgcn_s_wait_asynccnt)
  __builtin_amdgcn_s_wait_asynccnt(0);
#else
  asm volatile("s_wait_asynccnt 0x0" ::: "memory");
#endif
#endif
}

__device__ __forceinline__ bf16_t f32_to_bf16(float f) {
  unsigned u = __builtin_bit_cast(unsigned, f);
  unsigned r = u + 0x7FFFu + ((u >> 16) & 1u);   // round-to-nearest-even
  unsigned short h = (unsigned short)(r >> 16);
  return __builtin_bit_cast(bf16_t, h);
}

__device__ __forceinline__ v8f zero8() {
  v8f z;
#pragma unroll
  for (int i = 0; i < 8; ++i) z[i] = 0.0f;
  return z;
}

__device__ __forceinline__ v16bf pack16(v8bf lo, v8bf hi) {
  v16bf r;
#pragma unroll
  for (int i = 0; i < 8; ++i) { r[i] = lo[i]; r[i + 8] = hi[i]; }
  return r;
}

__device__ __forceinline__ v8f wmma_bf16(v16bf a, v16bf b, v8f c) {
  // D(16x16,f32) = A(16x32,bf16) * B(32x16,bf16) + C
  return __builtin_amdgcn_wmma_f32_16x16x32_bf16(false, a, false, b, (short)0, c,
                                                 false, false);
}

__device__ __forceinline__ void store_out(bf16_t* p, float v) { *p = f32_to_bf16(v); }
__device__ __forceinline__ void store_out(float* p, float v) { *p = v; }

// ---------------------------------------------------------------------------
// fp32 -> bf16 conversion, 4 elements per thread
// ---------------------------------------------------------------------------
__global__ void cvt_f32_to_bf16x4(const float* __restrict__ in,
                                  bf16_t* __restrict__ out, long n4) {
  long i = (long)blockIdx.x * blockDim.x + threadIdx.x;
  if (i < n4) {
    float4 f = ((const float4*)in)[i];
    bf16_t* p = out + i * 4;
    p[0] = f32_to_bf16(f.x);
    p[1] = f32_to_bf16(f.y);
    p[2] = f32_to_bf16(f.z);
    p[3] = f32_to_bf16(f.w);
  }
}

// ---------------------------------------------------------------------------
// Generic bf16 GEMM: C[M,N] = A[M,K] * B[K,N] + bias[N]
// 128x128 block tile, BK=32, 8 waves, each wave = 4x2 WMMA tiles (64x32)
// A tile staged via async global->LDS copies (ASYNCcnt), B tile staged
// transposed on the sync path.
// ---------------------------------------------------------------------------
template <typename OutT>
__global__ __launch_bounds__(256) void gemm_bf16(const bf16_t* __restrict__ A,
                                                 const bf16_t* __restrict__ Bm,
                                                 const float* __restrict__ bias,
                                                 OutT* __restrict__ C,
                                                 int M, int N, int K) {
  constexpr int BM = 128, BN = 128, BK = 32, BKP = 48;  // pad rows to 96B
  __shared__ bf16_t As[BM][BKP];
  __shared__ bf16_t Bs[BN][BKP];  // B tile stored transposed: Bs[n][k]

  const int tid = threadIdx.x;
  const int wave = tid >> 5, lane = tid & 31;
  const int lh = lane >> 4, lr = lane & 15;  // half-wave, lane-in-half
  const int wm = wave >> 2, wn = wave & 3;
  const int rowBase = blockIdx.y * BM;
  const int colBase = blockIdx.x * BN;
  const int wRow = wm * 64, wCol = wn * 32;

  // per-thread staging coordinates
  const int ar = tid >> 2;             // A tile row 0..63 (+64 for 2nd)
  const int ac = (tid & 3) * 8;        // A tile col chunk
  const int bn = tid & 127;            // B tile column
  const int bk = (tid >> 7) * 16;      // B tile k base (0 or 16)

  v8f acc[4][2];
#pragma unroll
  for (int i = 0; i < 4; ++i)
#pragma unroll
    for (int j = 0; j < 2; ++j) acc[i][j] = zero8();

  for (int k0 = 0; k0 < K; k0 += BK) {
    // prefetch next k-step tiles into L2/WGP$ while this step is staged
    if (k0 + BK < K) {
      __builtin_prefetch(A + (size_t)(rowBase + ar) * K + (k0 + BK + ac), 0, 1);
      __builtin_prefetch(Bm + (size_t)(k0 + BK + bk) * N + (colBase + bn), 0, 1);
    }
    {  // stage A tile: async global->LDS, 16B per lane x2
      const bf16_t* gA = A + (size_t)(rowBase + ar) * K + (k0 + ac);
      async_ld16(&As[ar][ac], gA);
      async_ld16(&As[ar + 64][ac], gA + (size_t)64 * K);
    }
    {  // stage B tile transposed: Bs[n][k] = B[k0+k][colBase+n] (sync path)
      const bf16_t* gB = Bm + (size_t)(k0 + bk) * N + (colBase + bn);
#pragma unroll
      for (int kk = 0; kk < 16; ++kk) Bs[bn][bk + kk] = gB[(size_t)kk * N];
    }
    wait_async();
    __syncthreads();

    // A fragment (16x32 bf16): lanes 0-15 hold K 0..7/16..23, lanes 16-31 +8
    v16bf af[4];
#pragma unroll
    for (int mt = 0; mt < 4; ++mt) {
      int row = wRow + mt * 16 + lr;
      v8bf lo = *(const v8bf*)&As[row][8 * lh];
      v8bf hi = *(const v8bf*)&As[row][16 + 8 * lh];
      af[mt] = pack16(lo, hi);
    }
    // B fragment (32x16 bf16): lanes 0-15 K 0..7/16..23, lanes 16-31 +8
    v16bf bfr[2];
#pragma unroll
    for (int nt = 0; nt < 2; ++nt) {
      int col = wCol + nt * 16 + lr;
      v8bf lo = *(const v8bf*)&Bs[col][8 * lh];
      v8bf hi = *(const v8bf*)&Bs[col][16 + 8 * lh];
      bfr[nt] = pack16(lo, hi);
    }
#pragma unroll
    for (int mt = 0; mt < 4; ++mt)
#pragma unroll
      for (int nt = 0; nt < 2; ++nt)
        acc[mt][nt] = wmma_bf16(af[mt], bfr[nt], acc[mt][nt]);
    __syncthreads();
  }

  // epilogue: D layout — VGPR r: lanes 0-15 M=r, lanes 16-31 M=r+8; N=lr
#pragma unroll
  for (int mt = 0; mt < 4; ++mt)
#pragma unroll
    for (int nt = 0; nt < 2; ++nt) {
      int col = colBase + wCol + nt * 16 + lr;
      float bv = bias[col];
#pragma unroll
      for (int r = 0; r < 8; ++r) {
        int row = rowBase + wRow + mt * 16 + lh * 8 + r;
        store_out(&C[(size_t)row * N + col], acc[mt][nt][r] + bv);
      }
    }
}

// ---------------------------------------------------------------------------
// Flash attention forward. qkv: [B*T, 3072] bf16 (Q|K|V, head h at h*64).
// Grid: (T/128, B*H). Block 256 = 8 waves; wave w owns query rows
// q0+16w .. q0+16w+15. Online softmax, key blocks of 64.
// K tile staged async; V tile staged transposed on sync path.
// out: [B*T, 1024] bf16, laid out as (b,t,h*64+d).
// ---------------------------------------------------------------------------
__global__ __launch_bounds__(256) void attn_fwd(const bf16_t* __restrict__ qkv,
                                                const unsigned char* __restrict__ msk,
                                                bf16_t* __restrict__ out, int T) {
  constexpr int Hn = 16, Eo = 1024, RS = 3072;
  __shared__ bf16_t Ks[64][72];       // K tile [j][d] -> B-frags for Q*K^T
  __shared__ bf16_t Vt[64][72];       // V tile transposed [d][j] -> B-frags for P*V
  __shared__ bf16_t Ps[8][16][72];    // per-wave P tile, D-layout -> A-layout relay

  const int tid = threadIdx.x;
  const int wave = tid >> 5, lane = tid & 31;
  const int lh = lane >> 4, lr = lane & 15;
  const int b = blockIdx.y / Hn, h = blockIdx.y % Hn;
  const int q0 = (int)blockIdx.x * 128;
  const int qw = q0 + wave * 16;

  const bf16_t* base = qkv + (size_t)b * T * RS + h * 64;

  // per-thread staging coordinates
  const int kr = tid >> 2, kc = (tid & 3) * 8;     // K tile
  const int vd = tid & 63, vk0 = (tid >> 6) * 16;  // V transpose

  // Q fragments (hd=64 -> two 16x32 A-frags), kept in registers
  v16bf qf[2];
  {
    const bf16_t* qp = base + (size_t)(qw + lr) * RS;
#pragma unroll
    for (int s = 0; s < 2; ++s) {
      v8bf lo = *(const v8bf*)(qp + s * 32 + 8 * lh);
      v8bf hi = *(const v8bf*)(qp + s * 32 + 16 + 8 * lh);
      qf[s] = pack16(lo, hi);
    }
  }

  float mrow[8], lsum[8];
  v8f o[4];
#pragma unroll
  for (int r = 0; r < 8; ++r) { mrow[r] = -1e30f; lsum[r] = 0.0f; }
#pragma unroll
  for (int nt = 0; nt < 4; ++nt) o[nt] = zero8();

  const int kbEnd = (q0 + 128) / 64;  // causal: keys up to q0+127
  for (int kb = 0; kb < kbEnd; ++kb) {
    const int j0 = kb * 64;
    // prefetch next key block while this one is staged/consumed
    if (kb + 1 < kbEnd) {
      __builtin_prefetch(base + 1024 + (size_t)(j0 + 64 + kr) * RS + kc, 0, 1);
      __builtin_prefetch(base + 2048 + (size_t)(j0 + 64 + vk0) * RS + vd, 0, 1);
    }
    {  // stage K tile [j][d]: async global->LDS
      const bf16_t* kp = base + 1024 + (size_t)(j0 + kr) * RS;
      async_ld16(&Ks[kr][kc], kp + kc);
      async_ld16(&Ks[kr][kc + 32], kp + kc + 32);
    }
    {  // stage V tile transposed [d][j] (sync path)
      const bf16_t* vp = base + 2048 + (size_t)(j0 + vk0) * RS + vd;
#pragma unroll
      for (int kk = 0; kk < 16; ++kk) Vt[vd][vk0 + kk] = vp[(size_t)kk * RS];
    }
    wait_async();
    __syncthreads();

    // S = Q * K^T * scale, 4 tiles of 16x16 over 64 keys
    v8f sc[4];
#pragma unroll
    for (int nt = 0; nt < 4; ++nt) {
      v8f s = zero8();
#pragma unroll
      for (int s2 = 0; s2 < 2; ++s2) {
        int col = nt * 16 + lr;
        v8bf lo = *(const v8bf*)&Ks[col][s2 * 32 + 8 * lh];
        v8bf hi = *(const v8bf*)&Ks[col][s2 * 32 + 16 + 8 * lh];
        s = wmma_bf16(qf[s2], pack16(lo, hi), s);
      }
      int j = j0 + nt * 16 + lr;
      bool ok = msk[(size_t)b * T + j] != 0;
#pragma unroll
      for (int r = 0; r < 8; ++r) {
        int i = qw + r + 8 * lh;
        float v = s[r] * 0.125f;  // 1/sqrt(64)
        sc[nt][r] = (j <= i && ok) ? v : -1e30f;
      }
    }

    // online softmax: rows live across 16-lane halves -> shfl_xor width 16
    float nm[8], alpha[8], psum[8];
#pragma unroll
    for (int r = 0; r < 8; ++r) {
      float v = fmaxf(fmaxf(sc[0][r], sc[1][r]), fmaxf(sc[2][r], sc[3][r]));
#pragma unroll
      for (int off = 8; off >= 1; off >>= 1) v = fmaxf(v, __shfl_xor(v, off, 16));
      nm[r] = fmaxf(mrow[r], v);
      alpha[r] = __expf(mrow[r] - nm[r]);
      mrow[r] = nm[r];
      lsum[r] *= alpha[r];
      psum[r] = 0.0f;
    }
#pragma unroll
    for (int nt = 0; nt < 4; ++nt) {
#pragma unroll
      for (int r = 0; r < 8; ++r) {
        o[nt][r] *= alpha[r];
        float p = __expf(sc[nt][r] - nm[r]);
        psum[r] += p;
        Ps[wave][r + 8 * lh][nt * 16 + lr] = f32_to_bf16(p);  // D-layout -> LDS
      }
    }
#pragma unroll
    for (int r = 0; r < 8; ++r) {
      float v = psum[r];
#pragma unroll
      for (int off = 8; off >= 1; off >>= 1) v += __shfl_xor(v, off, 16);
      lsum[r] += v;
    }

    // wave reads back its own P tile as A-fragments
    asm volatile("s_wait_dscnt 0x0" ::: "memory");
    v16bf pf[2];
#pragma unroll
    for (int s2 = 0; s2 < 2; ++s2) {
      v8bf lo = *(const v8bf*)&Ps[wave][lr][s2 * 32 + 8 * lh];
      v8bf hi = *(const v8bf*)&Ps[wave][lr][s2 * 32 + 16 + 8 * lh];
      pf[s2] = pack16(lo, hi);
    }
    // O += P * V  (K-dim = 64 keys, 4 hd tiles)
#pragma unroll
    for (int nt = 0; nt < 4; ++nt)
#pragma unroll
      for (int s2 = 0; s2 < 2; ++s2) {
        int d = nt * 16 + lr;
        v8bf lo = *(const v8bf*)&Vt[d][s2 * 32 + 8 * lh];
        v8bf hi = *(const v8bf*)&Vt[d][s2 * 32 + 16 + 8 * lh];
        o[nt] = wmma_bf16(pf[s2], pack16(lo, hi), o[nt]);
      }
    __syncthreads();
  }

  // normalize and store (b,t,h*64+d) bf16
#pragma unroll
  for (int nt = 0; nt < 4; ++nt) {
    int d = h * 64 + nt * 16 + lr;
#pragma unroll
    for (int r = 0; r < 8; ++r) {
      int t = qw + r + 8 * lh;
      out[(size_t)(b * T + t) * Eo + d] = f32_to_bf16(o[nt][r] / lsum[r]);
    }
  }
}

// ---------------------------------------------------------------------------
extern "C" void kernel_launch(void* const* d_in, const int* in_sizes, int n_in,
                              void* d_out, int out_size, void* d_ws, size_t ws_size,
                              hipStream_t stream) {
  const float* x            = (const float*)d_in[0];
  const unsigned char* mask = (const unsigned char*)d_in[1];
  const float* w_attn       = (const float*)d_in[2];
  const float* b_attn       = (const float*)d_in[3];
  const float* w_proj       = (const float*)d_in[4];
  const float* b_proj       = (const float*)d_in[5];
  float* out = (float*)d_out;

  constexpr int Bb = 2, T = 2048, E = 1024, Hh = 16;
  constexpr long M = (long)Bb * T;  // 4096

  // workspace layout (bf16): X | W_attn | W_proj | QKV | AttnOut  (~48 MB)
  bf16_t* Xb  = (bf16_t*)d_ws;
  bf16_t* Wab = Xb + M * E;
  bf16_t* Wpb = Wab + (long)E * 3 * E;
  bf16_t* QKV = Wpb + (long)E * E;
  bf16_t* AO  = QKV + M * 3 * E;

  auto cvt = [&](const float* src, bf16_t* dst, long n) {
    long n4 = n / 4;
    cvt_f32_to_bf16x4<<<dim3((unsigned)((n4 + 255) / 256)), dim3(256), 0, stream>>>(
        src, dst, n4);
  };
  cvt(x, Xb, M * E);
  cvt(w_attn, Wab, (long)E * 3 * E);
  cvt(w_proj, Wpb, (long)E * E);

  // QKV = X * W_attn + b_attn   (bf16 out)
  gemm_bf16<bf16_t><<<dim3(3 * E / 128, M / 128), dim3(256), 0, stream>>>(
      Xb, Wab, b_attn, QKV, (int)M, 3 * E, E);

  // flash attention
  attn_fwd<<<dim3(T / 128, Bb * Hh), dim3(256), 0, stream>>>(QKV, mask, AO, T);

  // out = AttnOut * W_proj + b_proj   (f32 out)
  gemm_bf16<float><<<dim3(E / 128, M / 128), dim3(256), 0, stream>>>(
      AO, Wpb, b_proj, out, (int)M, E, E);
}
